// GAT_3169685864885
// MI455X (gfx1250) — matrix operand order repaired and checked by reading, not compiled, not run on hardware
//
#include <hip/hip_runtime.h>
#include <hip/hip_bf16.h>
#include <math.h>

// ---------------------------------------------------------------------------
// Types for WMMA
// ---------------------------------------------------------------------------
typedef __attribute__((ext_vector_type(16))) _Float16 v16h;
typedef __attribute__((ext_vector_type(8)))  float    v8f;

union FragU {
  float4 f4[2];
  v16h   v;
};

// A fragment: per-lane halves are K offsets {base..base+7} and {base+16..base+23}
// (16-bit A 16x32 layout, ISA 7.12.2)
__device__ inline v16h load_fragA(const _Float16* p) {
  FragU u;
  u.f4[0] = *reinterpret_cast<const float4*>(p);
  u.f4[1] = *reinterpret_cast<const float4*>(p + 16);
  return u.v;
}
// B fragment: per-lane halves are 16 contiguous K values (contiguous k order, not the ISA table order; unused here)
__device__ inline v16h load_fragB(const _Float16* p) {
  FragU u;
  u.f4[0] = *reinterpret_cast<const float4*>(p);
  u.f4[1] = *reinterpret_cast<const float4*>(p + 8);
  return u.v;
}

// Async copy of 16 bytes global -> LDS (per lane). lds_off is the wave-relative
// LDS byte offset (low 32 bits of the generic pointer per the gfx1250 aperture
// mapping). Tracked by ASYNCcnt.
__device__ inline void async_copy16(unsigned lds_off, const void* gaddr) {
  asm volatile("global_load_async_to_lds_b128 %0, %1, off"
               :: "v"(lds_off), "v"(gaddr)
               : "memory");
}
__device__ inline void wait_async0() {
  asm volatile("s_wait_asynccnt 0x0" ::: "memory");
}

// ---------------------------------------------------------------------------
// Order-preserving float <-> uint mapping (for atomicMax-based segment max)
// ---------------------------------------------------------------------------
__device__ inline unsigned ordf(float f) {
  unsigned b = __float_as_uint(f);
  return (b & 0x80000000u) ? ~b : (b | 0x80000000u);
}
__device__ inline float unordf(unsigned u) {
  return (u & 0x80000000u) ? __uint_as_float(u & 0x7fffffffu)
                           : __uint_as_float(~u);
}

// ---------------------------------------------------------------------------
// f32 -> f16 conversion helpers (pre-pass so GEMM staging is a raw byte copy)
// ---------------------------------------------------------------------------
__global__ __launch_bounds__(256)
void conv_f16_kernel(const float* __restrict__ src, _Float16* __restrict__ dst, int total)
{
  int i = blockIdx.x * blockDim.x + threadIdx.x;
  if (i < total) dst[i] = (_Float16)src[i];
}

// Transpose + convert W[k][n] (k=0..127, n<wcols) into Wt[n][k] f16,
// zero-padding rows n in [wcols, nrows_pad).
__global__ __launch_bounds__(256)
void convW_kernel(const float* __restrict__ W, _Float16* __restrict__ Wt,
                  int wcols, int nrows_pad)
{
  int idx = blockIdx.x * blockDim.x + threadIdx.x;
  if (idx >= nrows_pad * 128) return;
  int n = idx >> 7;      // output row (= W column)
  int k = idx & 127;
  float v = (n < wcols) ? W[k * wcols + n] : 0.0f;
  Wt[idx] = (_Float16)v;
}

// ---------------------------------------------------------------------------
// WMMA GEMM: D[nrows x DCOLS] = Ah[nrows x 128] @ Wt^T, where
//   Ah : f16 row-major [nrows][128]
//   Wt : f16 row-major [>= gridDim.y*64][128]  (pre-transposed weight: Wt[n][k])
// block = 128 threads (4 waves). Each block: 16 rows x 64 cols.
// Tiles staged into LDS with the gfx1250 async-LDS pipe.
// ---------------------------------------------------------------------------
#define LDS_STRIDE 136  // halves; 272B row pitch (multiple of 16B)

template <int DCOLS>
__global__ __launch_bounds__(128)
void gemm_wmma_kernel(const _Float16* __restrict__ Ah, const _Float16* __restrict__ Wt,
                      float* __restrict__ D, int nrows)
{
  __shared__ __align__(16) _Float16 Alds[16 * LDS_STRIDE];
  __shared__ __align__(16) _Float16 Wlds[64 * LDS_STRIDE];

  const int t  = threadIdx.x;
  const int r0 = blockIdx.x * 16;
  const int n0 = blockIdx.y * 64;

  // ---- async stage A: 16 rows x (128 halves = 16 chunks of 16B) ----
  {
    unsigned lbase = (unsigned)(unsigned long long)(void*)Alds;
#pragma unroll
    for (int idx = t; idx < 16 * 16; idx += 128) {
      int m = idx >> 4, c = idx & 15;
      int row = r0 + m;
      if (row >= nrows) row = nrows - 1;   // clamp (N%16==0 so never hit here)
      const _Float16* g = Ah + (size_t)row * 128 + c * 8;
      unsigned l = lbase + (unsigned)(m * (LDS_STRIDE * 2) + c * 16);
      async_copy16(l, g);
    }
  }
  // ---- async stage W: 64 rows x 16 chunks of 16B ----
  {
    unsigned lbase = (unsigned)(unsigned long long)(void*)Wlds;
#pragma unroll
    for (int idx = t; idx < 64 * 16; idx += 128) {
      int n = idx >> 4, c = idx & 15;
      const _Float16* g = Wt + (size_t)(n0 + n) * 128 + c * 8;
      unsigned l = lbase + (unsigned)(n * (LDS_STRIDE * 2) + c * 16);
      async_copy16(l, g);
    }
  }
  wait_async0();
  __syncthreads();

  const int w = t >> 5;   // wave 0..3 -> 16-col subtile
  const int L = t & 31;
  const int n = L & 15;   // col (B/D) or row (A)
  const int g = L >> 4;   // half-wave group

  v8f acc = {};
#pragma unroll
  for (int kk = 0; kk < 128; kk += 32) {
    v16h af = load_fragA(&Alds[n * LDS_STRIDE + kk + 8 * g]);
    v16h bf = load_fragA(&Wlds[(w * 16 + n) * LDS_STRIDE + kk + 8 * g]);
    acc = __builtin_amdgcn_wmma_f32_16x16x32_f16(
        /*neg_a=*/false, af, /*neg_b=*/false, bf,
        /*c_mod=*/(short)0, acc, /*reuse_a=*/false, /*reuse_b=*/false);
  }

  const int colg = n0 + w * 16 + n;
  if (colg < DCOLS) {
    if (r0 + 15 < nrows) {   // full tile fast path (always taken: N%16==0)
      float* dp = D + (size_t)(r0 + 8 * g) * DCOLS + colg;
#pragma unroll
      for (int i = 0; i < 8; ++i) dp[i * DCOLS] = acc[i];
    } else {
#pragma unroll
      for (int i = 0; i < 8; ++i) {
        int row = r0 + i + 8 * g;
        if (row < nrows) D[(size_t)row * DCOLS + colg] = acc[i];
      }
    }
  }
}

// ---------------------------------------------------------------------------
// Per-(node,head) attention logits: as[n,h] = <h[n,h,:], att_src[h,:]>, same ad
// One wave per (node,head).
// ---------------------------------------------------------------------------
__global__ __launch_bounds__(256)
void alpha_kernel(const float* __restrict__ h, const float* __restrict__ att_s,
                  const float* __restrict__ att_d, float* __restrict__ as_out,
                  float* __restrict__ ad_out, int N, int H, int C)
{
  int wid = (int)((blockIdx.x * (size_t)blockDim.x + threadIdx.x) >> 5);
  int L = threadIdx.x & 31;
  if (wid >= N * H) return;
  int node = wid / H;
  int hh   = wid - node * H;
  const float* hp  = h + (size_t)node * H * C + (size_t)hh * C;
  const float* asp = att_s + hh * C;
  const float* adp = att_d + hh * C;
  float s = 0.0f, d = 0.0f;
  for (int c = L; c < C; c += 32) {
    float v = hp[c];
    s += v * asp[c];
    d += v * adp[c];
  }
#pragma unroll
  for (int off = 16; off > 0; off >>= 1) {
    s += __shfl_xor(s, off, 32);
    d += __shfl_xor(d, off, 32);
  }
  if (L == 0) { as_out[wid] = s; ad_out[wid] = d; }
}

// ---------------------------------------------------------------------------
// Edge softmax: segment max then segment sum (thread per edge; self-loops appended)
// ---------------------------------------------------------------------------
template <int H>
__global__ __launch_bounds__(256)
void edge_max_kernel(const int* __restrict__ ei, int E, int N,
                     const float* __restrict__ as, const float* __restrict__ ad,
                     unsigned* __restrict__ mbuf)
{
  int e = blockIdx.x * blockDim.x + threadIdx.x;
  if (e >= E + N) return;
  int s, d;
  if (e < E) { s = ei[e]; d = ei[E + e]; } else { s = d = e - E; }
#pragma unroll
  for (int h = 0; h < H; ++h) {
    float v = as[s * H + h] + ad[d * H + h];
    v = (v > 0.0f) ? v : 0.2f * v;   // leaky_relu
    atomicMax(&mbuf[d * H + h], ordf(v));
  }
}

template <int H>
__global__ __launch_bounds__(256)
void edge_sum_kernel(const int* __restrict__ ei, int E, int N,
                     const float* __restrict__ as, const float* __restrict__ ad,
                     const unsigned* __restrict__ mbuf, float* __restrict__ sbuf)
{
  int e = blockIdx.x * blockDim.x + threadIdx.x;
  if (e >= E + N) return;
  int s, d;
  if (e < E) { s = ei[e]; d = ei[E + e]; } else { s = d = e - E; }
#pragma unroll
  for (int h = 0; h < H; ++h) {
    float v = as[s * H + h] + ad[d * H + h];
    v = (v > 0.0f) ? v : 0.2f * v;
    float m = unordf(mbuf[d * H + h]);
    atomicAdd(&sbuf[d * H + h], __expf(v - m));
  }
}

// ---------------------------------------------------------------------------
// Layer-1 messages: wave per edge, H=2, C=64 (128 floats), lane -> 4 channels
// ---------------------------------------------------------------------------
__global__ __launch_bounds__(256)
void message1_kernel(const int* __restrict__ ei, int E, int N,
                     const float* __restrict__ as, const float* __restrict__ ad,
                     const unsigned* __restrict__ mbuf, const float* __restrict__ sbuf,
                     const float* __restrict__ h1, float* __restrict__ agg)
{
  int wid = (int)((blockIdx.x * (size_t)blockDim.x + threadIdx.x) >> 5);
  int L = threadIdx.x & 31;
  if (wid >= E + N) return;
  int s, d;
  if (wid < E) { s = ei[wid]; d = ei[E + wid]; } else { s = d = wid - E; }
  int c0 = L * 4;          // channel offset 0..124
  int h  = L >> 4;         // head = c0 / 64
  __builtin_prefetch(&h1[(size_t)s * 128 + c0], 0, 0);
  float v = as[s * 2 + h] + ad[d * 2 + h];
  v = (v > 0.0f) ? v : 0.2f * v;
  float m   = unordf(mbuf[d * 2 + h]);
  float sum = sbuf[d * 2 + h];
  float coef = __expf(v - m) / (sum + 1e-16f);
  float4 x4 = *reinterpret_cast<const float4*>(&h1[(size_t)s * 128 + c0]);
  float* ap = &agg[(size_t)d * 128 + c0];
  atomicAdd(ap + 0, x4.x * coef);
  atomicAdd(ap + 1, x4.y * coef);
  atomicAdd(ap + 2, x4.z * coef);
  atomicAdd(ap + 3, x4.w * coef);
}

// ---------------------------------------------------------------------------
// Layer-1 epilogue: x2h = f16(elu(agg1 + b1))   (feeds the async-staged GEMM2)
// ---------------------------------------------------------------------------
__global__ __launch_bounds__(256)
void epilogue1_kernel(const float* __restrict__ agg, const float* __restrict__ b1,
                      _Float16* __restrict__ x2h, int total)
{
  int i = blockIdx.x * blockDim.x + threadIdx.x;
  if (i >= total) return;
  float v = agg[i] + b1[i & 127];
  v = (v > 0.0f) ? v : expm1f(v);
  x2h[i] = (_Float16)v;
}

// ---------------------------------------------------------------------------
// Layer-2 messages: wave per edge, H=1, C=40
// ---------------------------------------------------------------------------
__global__ __launch_bounds__(256)
void message2_kernel(const int* __restrict__ ei, int E, int N,
                     const float* __restrict__ as, const float* __restrict__ ad,
                     const unsigned* __restrict__ mbuf, const float* __restrict__ sbuf,
                     const float* __restrict__ h2, float* __restrict__ agg)
{
  int wid = (int)((blockIdx.x * (size_t)blockDim.x + threadIdx.x) >> 5);
  int L = threadIdx.x & 31;
  if (wid >= E + N) return;
  int s, d;
  if (wid < E) { s = ei[wid]; d = ei[E + wid]; } else { s = d = wid - E; }
  float v = as[s] + ad[d];
  v = (v > 0.0f) ? v : 0.2f * v;
  float m   = unordf(mbuf[d]);
  float sum = sbuf[d];
  float coef = __expf(v - m) / (sum + 1e-16f);
  const float* hp = &h2[(size_t)s * 40];
  float* ap = &agg[(size_t)d * 40];
  for (int c = L; c < 40; c += 32) {
    atomicAdd(ap + c, hp[c] * coef);
  }
}

// ---------------------------------------------------------------------------
// Final: out = log_softmax(agg2 + b2), wave per node, 40 classes
// ---------------------------------------------------------------------------
__global__ __launch_bounds__(256)
void final_kernel(const float* __restrict__ agg, const float* __restrict__ b2,
                  float* __restrict__ out, int N)
{
  int n = (int)((blockIdx.x * (size_t)blockDim.x + threadIdx.x) >> 5);
  int L = threadIdx.x & 31;
  if (n >= N) return;
  const float* ap = &agg[(size_t)n * 40];
  float v0 = ap[L] + b2[L];
  float v1 = (L < 8) ? (ap[L + 32] + b2[L + 32]) : -INFINITY;
  float mx = fmaxf(v0, v1);
#pragma unroll
  for (int off = 16; off > 0; off >>= 1) mx = fmaxf(mx, __shfl_xor(mx, off, 32));
  float se = __expf(v0 - mx) + ((L < 8) ? __expf(v1 - mx) : 0.0f);
#pragma unroll
  for (int off = 16; off > 0; off >>= 1) se += __shfl_xor(se, off, 32);
  float lse = mx + logf(se);
  out[(size_t)n * 40 + L] = v0 - lse;
  if (L < 8) out[(size_t)n * 40 + L + 32] = v1 - lse;
}

// ---------------------------------------------------------------------------
// Launcher
// ---------------------------------------------------------------------------
extern "C" void kernel_launch(void* const* d_in, const int* in_sizes, int n_in,
                              void* d_out, int out_size, void* d_ws, size_t ws_size,
                              hipStream_t stream)
{
  const float* x        = (const float*)d_in[0];
  const int*   ei       = (const int*)d_in[1];
  const float* W1       = (const float*)d_in[2];
  const float* att_src1 = (const float*)d_in[3];
  const float* att_dst1 = (const float*)d_in[4];
  const float* b1       = (const float*)d_in[5];
  const float* W2       = (const float*)d_in[6];
  const float* att_src2 = (const float*)d_in[7];
  const float* att_dst2 = (const float*)d_in[8];
  const float* b2       = (const float*)d_in[9];
  float* out = (float*)d_out;

  const int N = in_sizes[0] / 128;   // 100000
  const int E = in_sizes[1] / 2;     // 1600000
  const int Etot = E + N;            // with self-loops

  // Workspace layout (256B aligned regions)
  char* ws = (char*)d_ws;
  size_t off = 0;
  auto take = [&](size_t bytes) -> char* {
    char* p = ws + off;
    off = (off + bytes + 255) & ~(size_t)255;
    return p;
  };
  float*     h1    = (float*)    take((size_t)N * 128 * 4);
  float*     agg1  = (float*)    take((size_t)N * 128 * 4);
  _Float16*  xh    = (_Float16*) take((size_t)N * 128 * 2);  // f16 input activations
  _Float16*  x2h   = (_Float16*) take((size_t)N * 128 * 2);  // f16 layer-2 activations
  _Float16*  wt1h  = (_Float16*) take((size_t)128 * 128 * 2); // W1^T f16
  _Float16*  wt2h  = (_Float16*) take((size_t)64 * 128 * 2);  // W2^T f16, padded to 64 rows
  float*     as1   = (float*)    take((size_t)N * 2 * 4);
  float*     ad1   = (float*)    take((size_t)N * 2 * 4);
  unsigned*  mbuf1 = (unsigned*) take((size_t)N * 2 * 4);
  float*     sbuf1 = (float*)    take((size_t)N * 2 * 4);
  float*     h2    = (float*)    take((size_t)N * 40 * 4);
  float*     agg2  = (float*)    take((size_t)N * 40 * 4);
  float*     as2   = (float*)    take((size_t)N * 4);
  float*     ad2   = (float*)    take((size_t)N * 4);
  unsigned*  mbuf2 = (unsigned*) take((size_t)N * 4);
  float*     sbuf2 = (float*)    take((size_t)N * 4);

  // Zero-init accumulators and softmax state (ordered-uint 0 == -inf)
  hipMemsetAsync(agg1,  0, (size_t)N * 128 * 4, stream);
  hipMemsetAsync(agg2,  0, (size_t)N * 40 * 4, stream);
  hipMemsetAsync(mbuf1, 0, (size_t)N * 2 * 4, stream);
  hipMemsetAsync(sbuf1, 0, (size_t)N * 2 * 4, stream);
  hipMemsetAsync(mbuf2, 0, (size_t)N * 4, stream);
  hipMemsetAsync(sbuf2, 0, (size_t)N * 4, stream);

  const int nrt    = (N + 15) / 16;              // 6250 row tiles
  const int egrid  = (Etot + 255) / 256;         // thread-per-edge kernels
  const int ewgrid = (Etot + 7) / 8;             // wave-per-edge kernels (8 waves/block)

  // f16 pre-conversion (x, W1^T, W2^T)
  conv_f16_kernel<<<(N * 128 + 255) / 256, 256, 0, stream>>>(x, xh, N * 128);
  convW_kernel<<<(128 * 128 + 255) / 256, 256, 0, stream>>>(W1, wt1h, 128, 128);
  convW_kernel<<<(64 * 128 + 255) / 256, 256, 0, stream>>>(W2, wt2h, 40, 64);

  // ---- Layer 1 ----
  gemm_wmma_kernel<128><<<dim3(nrt, 2), 128, 0, stream>>>(xh, wt1h, h1, N);
  alpha_kernel<<<(N * 2 * 32 + 255) / 256, 256, 0, stream>>>(
      h1, att_src1, att_dst1, as1, ad1, N, 2, 64);
  edge_max_kernel<2><<<egrid, 256, 0, stream>>>(ei, E, N, as1, ad1, mbuf1);
  edge_sum_kernel<2><<<egrid, 256, 0, stream>>>(ei, E, N, as1, ad1, mbuf1, sbuf1);
  message1_kernel<<<ewgrid, 256, 0, stream>>>(ei, E, N, as1, ad1, mbuf1, sbuf1, h1, agg1);
  epilogue1_kernel<<<(N * 128 + 255) / 256, 256, 0, stream>>>(agg1, b1, x2h, N * 128);

  // ---- Layer 2 ----
  gemm_wmma_kernel<40><<<dim3(nrt, 1), 128, 0, stream>>>(x2h, wt2h, h2, N);
  alpha_kernel<<<(N * 32 + 255) / 256, 256, 0, stream>>>(
      h2, att_src2, att_dst2, as2, ad2, N, 1, 40);
  edge_max_kernel<1><<<egrid, 256, 0, stream>>>(ei, E, N, as2, ad2, mbuf2);
  edge_sum_kernel<1><<<egrid, 256, 0, stream>>>(ei, E, N, as2, ad2, mbuf2, sbuf2);
  message2_kernel<<<ewgrid, 256, 0, stream>>>(ei, E, N, as2, ad2, mbuf2, sbuf2, h2, agg2);
  final_kernel<<<(N * 32 + 255) / 256, 256, 0, stream>>>(agg2, b2, out, N);
}
